// Seq2Seq_42717744726467
// MI455X (gfx1250) — matrix operand order repaired and checked by reading, not compiled, not run on hardware
//
#include <hip/hip_runtime.h>
#include <hip/hip_bf16.h>

typedef __bf16 bf16_t;
typedef __attribute__((ext_vector_type(16))) __bf16 v16bf;
typedef __attribute__((ext_vector_type(8)))  float  v8f;

#define V_SIZE  32000
#define E_DIM   256
#define H_DIM   512
#define SRC_LEN 400
#define TGT_LEN 100
#define B_SIZE  32
#define SOS_TOK 1

// D = A*B + C with bf16 operands, f32 accumulate
#define WMMA_BF16(acc, a, b)                                                   \
  acc = __builtin_amdgcn_wmma_f32_16x16x32_bf16(false, (a), false, (b),        \
                                                (short)0, (acc), false, false)

// ---- CDNA5 16-bit A-matrix 16x32 layout (CDNA5 ISA):
// lanes 0-15: M=lane, elements 0..7 = K(0..7), 8..15 = K(16..23)
// lanes 16-31: M=lane-16, elements 0..7 = K(8..15), 8..15 = K(24..31)
static __device__ __forceinline__ v16bf load_A_tile(const bf16_t* rowk0, int lane) {
  const int koff = (lane >> 4) << 3; // 0 or 8
  v16bf a;
  ((float4*)&a)[0] = *(const float4*)(rowk0 + koff);
  ((float4*)&a)[1] = *(const float4*)(rowk0 + 16 + koff);
  return a;
}
// ---- 16-bit B-matrix 32x16 layout: lane n (0-15): N=n, elements 0..7 = K(0..7), 8..15 = K(16..23);
// lane 16+n: N=n, elements 0..7 = K(8..15), 8..15 = K(24..31).  (colk0 points at activation row + k0)
static __device__ __forceinline__ v16bf load_B_tile(const bf16_t* colk0, int lane) {
  const int koff = (lane >> 4) << 3; // 0 or 8
  v16bf b;
  ((float4*)&b)[0] = *(const float4*)(colk0 + koff);
  ((float4*)&b)[1] = *(const float4*)(colk0 + 16 + koff);
  return b;
}

static __device__ __forceinline__ float sigm(float x) {
  return 1.0f / (1.0f + __expf(-x));
}

// --------------------------------------------------------------------------
// fp32 -> bf16 bulk conversion (weights become L2-resident bf16)
__global__ void f32_to_bf16_kernel(const float* __restrict__ src,
                                   bf16_t* __restrict__ dst, size_t n) {
  size_t i = (size_t)blockIdx.x * blockDim.x + threadIdx.x;
  const size_t stride = (size_t)gridDim.x * blockDim.x;
  for (; i < n; i += stride) dst[i] = (bf16_t)src[i];
}

// zero h0 (fp32 + bf16 mirror) and the total-loss slot
__global__ void init_kernel(float* h0_f, bf16_t* h0_bf, float* loss_slot) {
  const int i = blockIdx.x * blockDim.x + threadIdx.x;
  if (i < B_SIZE * H_DIM) { h0_f[i] = 0.0f; h0_bf[i] = (bf16_t)0.0f; }
  if (i == 0) *loss_slot = 0.0f;
}

// --------------------------------------------------------------------------
// One GRU step.  grid = H/16 = 32 blocks, blockDim = 32 (one wave).
// Block b computes h columns [16b, 16b+16) for all 32 batches.
// D[m = h-col][n = batch];  A = weight rows (row-major), B = activations
// with lane = batch.  Both batch halves processed inside the K loop so each
// A (weight) tile is loaded once and feeds two WMMAs — no outer-loop spills.
__global__ __launch_bounds__(32, 1)
void gru_step_kernel(const bf16_t* __restrict__ Wih_bf,  // (3H,E)
                     const bf16_t* __restrict__ Whh_bf,  // (3H,H)
                     const float*  __restrict__ b_ih,    // (3H)
                     const float*  __restrict__ b_hh,    // (3H)
                     const bf16_t* __restrict__ emb_bf,  // (V,E)
                     const int*    __restrict__ tok_row, // B tokens or null
                     int const_tok,
                     const float*  __restrict__ h_in,    // (B,H)
                     const bf16_t* __restrict__ h_in_bf, // (B,H)
                     float*        __restrict__ h_out,
                     bf16_t*       __restrict__ h_out_bf) {
  const int lane  = threadIdx.x;
  const int hbase = blockIdx.x * 16;
  const int mrow  = lane & 15;

  // A-operand row pointers for the three gates (PyTorch order r,z,n)
  const bf16_t* Air = Wih_bf + (size_t)(            hbase + mrow) * E_DIM;
  const bf16_t* Aiz = Wih_bf + (size_t)(    H_DIM + hbase + mrow) * E_DIM;
  const bf16_t* Ain = Wih_bf + (size_t)(2 * H_DIM + hbase + mrow) * E_DIM;
  const bf16_t* Ahr = Whh_bf + (size_t)(            hbase + mrow) * H_DIM;
  const bf16_t* Ahz = Whh_bf + (size_t)(    H_DIM + hbase + mrow) * H_DIM;
  const bf16_t* Ahn = Whh_bf + (size_t)(2 * H_DIM + hbase + mrow) * H_DIM;

  // B-operand row pointers for both batch halves this lane serves
  const int bb0 = mrow, bb1 = 16 + mrow;
  const int tok0 = tok_row ? tok_row[bb0] : const_tok;
  const int tok1 = tok_row ? tok_row[bb1] : const_tok;
  const bf16_t* xrow0 = emb_bf  + (size_t)tok0 * E_DIM;
  const bf16_t* xrow1 = emb_bf  + (size_t)tok1 * E_DIM;
  const bf16_t* hrow0 = h_in_bf + (size_t)bb0  * H_DIM;
  const bf16_t* hrow1 = h_in_bf + (size_t)bb1  * H_DIM;

  v8f r0 = {}, z0 = {}, gin0 = {}, ghn0 = {};
  v8f r1 = {}, z1 = {}, gin1 = {}, ghn1 = {};

  // gi contributions (K over E); r/z accumulate gi+gh into one C chain
  for (int k0 = 0; k0 < E_DIM; k0 += 32) {
    const v16bf b0 = load_B_tile(xrow0 + k0, lane);
    const v16bf b1 = load_B_tile(xrow1 + k0, lane);
    const v16bf ar = load_A_tile(Air + k0, lane);
    const v16bf az = load_A_tile(Aiz + k0, lane);
    const v16bf an = load_A_tile(Ain + k0, lane);
    WMMA_BF16(r0,   ar, b0);  WMMA_BF16(r1,   ar, b1);
    WMMA_BF16(z0,   az, b0);  WMMA_BF16(z1,   az, b1);
    WMMA_BF16(gin0, an, b0);  WMMA_BF16(gin1, an, b1);
  }
  // gh contributions (K over H); gh_n kept separate (needed after r gating)
  for (int k0 = 0; k0 < H_DIM; k0 += 32) {
    const v16bf b0 = load_B_tile(hrow0 + k0, lane);
    const v16bf b1 = load_B_tile(hrow1 + k0, lane);
    const v16bf ar = load_A_tile(Ahr + k0, lane);
    const v16bf az = load_A_tile(Ahz + k0, lane);
    const v16bf an = load_A_tile(Ahn + k0, lane);
    WMMA_BF16(r0,   ar, b0);  WMMA_BF16(r1,   ar, b1);
    WMMA_BF16(z0,   az, b0);  WMMA_BF16(z1,   az, b1);
    WMMA_BF16(ghn0, an, b0);  WMMA_BF16(ghn1, an, b1);
  }

  // D layout: VGPR e holds M = e (lanes 0-15) or 8+e (lanes 16-31), N = mrow
  const int mofs = (lane < 16) ? 0 : 8;
#pragma unroll
  for (int half = 0; half < 2; ++half) {
    const int bb = half ? bb1 : bb0;
    const v8f& ra = half ? r1 : r0;
    const v8f& za = half ? z1 : z0;
    const v8f& ga = half ? gin1 : gin0;
    const v8f& ha = half ? ghn1 : ghn0;
#pragma unroll
    for (int e = 0; e < 8; ++e) {
      const int j = hbase + mofs + e;
      const float rg = sigm(ra[e] + b_ih[j] + b_hh[j]);
      const float zg = sigm(za[e] + b_ih[H_DIM + j] + b_hh[H_DIM + j]);
      const float nn = tanhf(ga[e] + b_ih[2 * H_DIM + j] +
                             rg * (ha[e] + b_hh[2 * H_DIM + j]));
      const float ho = h_in[(size_t)bb * H_DIM + j];
      const float hn = (1.0f - zg) * nn + zg * ho;
      h_out   [(size_t)bb * H_DIM + j] = hn;
      h_out_bf[(size_t)bb * H_DIM + j] = (bf16_t)hn;
    }
  }
}

// --------------------------------------------------------------------------
// out_emb = h @ pre_W^T + pre_b.  grid = E/16 = 16 blocks, blockDim = 32.
__global__ __launch_bounds__(32, 1)
void pre_proj_kernel(const bf16_t* __restrict__ preW_bf, // (E,H)
                     const float*  __restrict__ pre_b,   // (E)
                     const bf16_t* __restrict__ h_bf,    // (B,H)
                     float*        __restrict__ out_emb, // (B,E)
                     bf16_t*       __restrict__ out_emb_bf) {
  const int lane  = threadIdx.x;
  const int ebase = blockIdx.x * 16;
  const int mrow  = lane & 15;
  const bf16_t* Arow  = preW_bf + (size_t)(ebase + mrow) * H_DIM;
  const bf16_t* hrow0 = h_bf + (size_t)mrow * H_DIM;
  const bf16_t* hrow1 = h_bf + (size_t)(16 + mrow) * H_DIM;

  v8f acc0 = {}, acc1 = {};
  for (int k0 = 0; k0 < H_DIM; k0 += 32) {
    const v16bf a  = load_A_tile(Arow + k0, lane);
    const v16bf b0 = load_B_tile(hrow0 + k0, lane);
    const v16bf b1 = load_B_tile(hrow1 + k0, lane);
    WMMA_BF16(acc0, a, b0);
    WMMA_BF16(acc1, a, b1);
  }
  const int mofs = (lane < 16) ? 0 : 8;
#pragma unroll
  for (int e = 0; e < 8; ++e) {
    const int j = ebase + mofs + e;
    const float v0 = acc0[e] + pre_b[j];
    const float v1 = acc1[e] + pre_b[j];
    out_emb   [(size_t)mrow * E_DIM + j]        = v0;
    out_emb_bf[(size_t)mrow * E_DIM + j]        = (bf16_t)v0;
    out_emb   [(size_t)(16 + mrow) * E_DIM + j] = v1;
    out_emb_bf[(size_t)(16 + mrow) * E_DIM + j] = (bf16_t)v1;
  }
}

// --------------------------------------------------------------------------
// logits = out_emb @ embedding^T + out_b (tied weights).
// grid = V/16 = 2000 blocks, blockDim = 32.  logits stored (B,V) row-major.
__global__ __launch_bounds__(32, 1)
void logits_kernel(const bf16_t* __restrict__ emb_bf,   // (V,E)
                   const float*  __restrict__ out_b,    // (V)
                   const bf16_t* __restrict__ oemb_bf,  // (B,E)
                   float*        __restrict__ logits) { // (B,V)
  const int lane  = threadIdx.x;
  const int vbase = blockIdx.x * 16;
  const int mrow  = lane & 15;
  const bf16_t* Arow  = emb_bf + (size_t)(vbase + mrow) * E_DIM;
  const bf16_t* brow0 = oemb_bf + (size_t)mrow * E_DIM;
  const bf16_t* brow1 = oemb_bf + (size_t)(16 + mrow) * E_DIM;
  __builtin_prefetch(Arow, 0, 1);  // global_prefetch_b8 warm-up of the A row

  v8f acc0 = {}, acc1 = {};
#pragma unroll
  for (int k0 = 0; k0 < E_DIM; k0 += 32) {
    const v16bf a  = load_A_tile(Arow + k0, lane);
    const v16bf b0 = load_B_tile(brow0 + k0, lane);
    const v16bf b1 = load_B_tile(brow1 + k0, lane);
    WMMA_BF16(acc0, a, b0);
    WMMA_BF16(acc1, a, b1);
  }
  const int mofs = (lane < 16) ? 0 : 8;
#pragma unroll
  for (int e = 0; e < 8; ++e) {
    const int vc = vbase + mofs + e;
    const float bias = out_b[vc];
    logits[(size_t)mrow * V_SIZE + vc]        = acc0[e] + bias;
    logits[(size_t)(16 + mrow) * V_SIZE + vc] = acc1[e] + bias;
  }
}

// --------------------------------------------------------------------------
// Per-batch softmax + argmax + masked NLL.  grid = B = 32, blockDim = 256.
__global__ void softmax_loss_kernel(const float* __restrict__ logits,  // (B,V)
                                    const int*   __restrict__ gold_row,// (B)
                                    float*       __restrict__ tokens_out,
                                    float*       __restrict__ nll_buf,
                                    float*       __restrict__ mask_buf) {
  const int n   = blockIdx.x;
  const int tid = threadIdx.x;
  const float* row = logits + (size_t)n * V_SIZE;
  __shared__ float smax[256];
  __shared__ int   sidx[256];
  __shared__ float ssum[256];

  float bmax = -3.0e38f; int bidx = 0;
  for (int v = tid; v < V_SIZE; v += 256) {
    const float x = row[v];
    if (x > bmax) { bmax = x; bidx = v; }     // strided: keeps earliest index
  }
  smax[tid] = bmax; sidx[tid] = bidx;
  __syncthreads();
  for (int s = 128; s > 0; s >>= 1) {
    if (tid < s) {
      const float o = smax[tid + s]; const int oi = sidx[tid + s];
      if (o > smax[tid] || (o == smax[tid] && oi < sidx[tid])) {
        smax[tid] = o; sidx[tid] = oi;        // jnp.argmax: first max wins
      }
    }
    __syncthreads();
  }
  const float mx = smax[0];
  float lsum = 0.0f;
  for (int v = tid; v < V_SIZE; v += 256) lsum += __expf(row[v] - mx);
  ssum[tid] = lsum;
  __syncthreads();
  for (int s = 128; s > 0; s >>= 1) {
    if (tid < s) ssum[tid] += ssum[tid + s];
    __syncthreads();
  }
  if (tid == 0) {
    tokens_out[n] = (float)sidx[0];
    const int g = gold_row[n];
    const float p = __expf(row[g] - mx) / ssum[0];
    const float nll = -logf(p + 1e-20f);
    const float m = (g != 0) ? 1.0f : 0.0f;   // PAD == 0
    nll_buf[n]  = nll * m;
    mask_buf[n] = m;
  }
}

// masked mean over batch, accumulated into total loss (serial per-step)
__global__ void finalize_loss_kernel(const float* __restrict__ nll_buf,
                                     const float* __restrict__ mask_buf,
                                     float* __restrict__ total_loss) {
  if (threadIdx.x == 0 && blockIdx.x == 0) {
    float sn = 0.0f, sm = 0.0f;
    for (int i = 0; i < B_SIZE; ++i) { sn += nll_buf[i]; sm += mask_buf[i]; }
    *total_loss += sn / fmaxf(sm, 1.0f);
  }
}

// --------------------------------------------------------------------------
extern "C" void kernel_launch(void* const* d_in, const int* in_sizes, int n_in,
                              void* d_out, int out_size, void* d_ws, size_t ws_size,
                              hipStream_t stream) {
  const int*   src  = (const int*)  d_in[0];
  const int*   tgt  = (const int*)  d_in[1];
  const float* emb  = (const float*)d_in[2];
  const float* eWih = (const float*)d_in[3];
  const float* eWhh = (const float*)d_in[4];
  const float* eBih = (const float*)d_in[5];
  const float* eBhh = (const float*)d_in[6];
  const float* dWih = (const float*)d_in[7];
  const float* dWhh = (const float*)d_in[8];
  const float* dBih = (const float*)d_in[9];
  const float* dBhh = (const float*)d_in[10];
  const float* preW = (const float*)d_in[11];
  const float* preB = (const float*)d_in[12];
  const float* outB = (const float*)d_in[13];
  float* out = (float*)d_out;   // [TGT_LEN*B tokens (as float), 1 total loss]

  // ---- workspace carve-up (256B aligned) ----
  char* p = (char*)d_ws;
  auto carve = [&](size_t bytes) {
    char* r = p;
    p += (bytes + 255) & ~(size_t)255;
    return (void*)r;
  };
  bf16_t* emb_bf  = (bf16_t*)carve((size_t)V_SIZE * E_DIM * sizeof(bf16_t));
  bf16_t* eWih_bf = (bf16_t*)carve((size_t)3 * H_DIM * E_DIM * sizeof(bf16_t));
  bf16_t* eWhh_bf = (bf16_t*)carve((size_t)3 * H_DIM * H_DIM * sizeof(bf16_t));
  bf16_t* dWih_bf = (bf16_t*)carve((size_t)3 * H_DIM * E_DIM * sizeof(bf16_t));
  bf16_t* dWhh_bf = (bf16_t*)carve((size_t)3 * H_DIM * H_DIM * sizeof(bf16_t));
  bf16_t* preW_bf = (bf16_t*)carve((size_t)E_DIM * H_DIM * sizeof(bf16_t));
  float*  h_f[2]  = {(float*)carve((size_t)B_SIZE * H_DIM * 4),
                     (float*)carve((size_t)B_SIZE * H_DIM * 4)};
  bf16_t* h_b[2]  = {(bf16_t*)carve((size_t)B_SIZE * H_DIM * 2),
                     (bf16_t*)carve((size_t)B_SIZE * H_DIM * 2)};
  float*  oemb_f  = (float*) carve((size_t)B_SIZE * E_DIM * 4);
  bf16_t* oemb_b  = (bf16_t*)carve((size_t)B_SIZE * E_DIM * 2);
  float*  logits  = (float*) carve((size_t)B_SIZE * V_SIZE * 4);
  float*  nll_buf = (float*) carve(B_SIZE * 4);
  float*  mask_buf= (float*) carve(B_SIZE * 4);

  // ---- one-time-per-call weight conversion to bf16 (L2-resident after) ----
  f32_to_bf16_kernel<<<2048, 256, 0, stream>>>(emb,  emb_bf,  (size_t)V_SIZE * E_DIM);
  f32_to_bf16_kernel<<<512,  256, 0, stream>>>(eWih, eWih_bf, (size_t)3 * H_DIM * E_DIM);
  f32_to_bf16_kernel<<<512,  256, 0, stream>>>(eWhh, eWhh_bf, (size_t)3 * H_DIM * H_DIM);
  f32_to_bf16_kernel<<<512,  256, 0, stream>>>(dWih, dWih_bf, (size_t)3 * H_DIM * E_DIM);
  f32_to_bf16_kernel<<<512,  256, 0, stream>>>(dWhh, dWhh_bf, (size_t)3 * H_DIM * H_DIM);
  f32_to_bf16_kernel<<<256,  256, 0, stream>>>(preW, preW_bf, (size_t)E_DIM * H_DIM);

  init_kernel<<<(B_SIZE * H_DIM + 255) / 256, 256, 0, stream>>>(
      h_f[0], h_b[0], out + (size_t)TGT_LEN * B_SIZE);

  // ---- encoder: 400 sequential GRU steps (double-buffered hidden state) ----
  int s = 0;
  for (int t = 0; t < SRC_LEN; ++t, ++s) {
    const int in = s & 1, ot = in ^ 1;
    gru_step_kernel<<<H_DIM / 16, 32, 0, stream>>>(
        eWih_bf, eWhh_bf, eBih, eBhh, emb_bf,
        src + (size_t)t * B_SIZE, 0,
        h_f[in], h_b[in], h_f[ot], h_b[ot]);
  }

  // ---- decoder with teacher forcing: 100 steps ----
  for (int t = 0; t < TGT_LEN; ++t, ++s) {
    const int in = s & 1, ot = in ^ 1;
    const int* tok_row = (t == 0) ? (const int*)nullptr
                                  : tgt + (size_t)(t - 1) * B_SIZE;
    gru_step_kernel<<<H_DIM / 16, 32, 0, stream>>>(
        dWih_bf, dWhh_bf, dBih, dBhh, emb_bf,
        tok_row, SOS_TOK,
        h_f[in], h_b[in], h_f[ot], h_b[ot]);
    pre_proj_kernel<<<E_DIM / 16, 32, 0, stream>>>(
        preW_bf, preB, h_b[ot], oemb_f, oemb_b);
    logits_kernel<<<V_SIZE / 16, 32, 0, stream>>>(
        emb_bf, outB, oemb_b, logits);
    softmax_loss_kernel<<<B_SIZE, 256, 0, stream>>>(
        logits, tgt + (size_t)t * B_SIZE,
        out + (size_t)t * B_SIZE, nll_buf, mask_buf);
    finalize_loss_kernel<<<1, 32, 0, stream>>>(
        nll_buf, mask_buf, out + (size_t)TGT_LEN * B_SIZE);
  }
  (void)in_sizes; (void)n_in; (void)out_size; (void)ws_size;
}
